// Attention_33122787787435
// MI455X (gfx1250) — compile-verified, matrix operand order checked by reading, not run on hardware
//
#include <hip/hip_runtime.h>

// ---------------------------------------------------------------------------
// Types / target feature probes
// ---------------------------------------------------------------------------
typedef __attribute__((ext_vector_type(16))) __bf16 v16bf;
typedef __attribute__((ext_vector_type(8)))  __bf16 v8bf;
typedef __attribute__((ext_vector_type(4)))  __bf16 v4bf;
typedef __attribute__((ext_vector_type(8)))  float  v8f;
typedef int v4i __attribute__((vector_size(16)));   // matches builtin param

#define AS1 __attribute__((address_space(1)))
#define AS3 __attribute__((address_space(3)))

#if __has_builtin(__builtin_amdgcn_global_load_async_to_lds_b128)
#define HAVE_ASYNC_LDS 1
#else
#define HAVE_ASYNC_LDS 0
#endif

__device__ __forceinline__ v8f wmma_bf16(v16bf a, v16bf b, v8f c) {
    // v_wmma_f32_16x16x32_bf16  (D = A x B + C), wave32
    return __builtin_amdgcn_wmma_f32_16x16x32_bf16(
        /*neg_a=*/false, a, /*neg_b=*/false, b,
        /*c_mod=*/(short)0, c, /*reuse_a=*/false, /*reuse_b=*/false);
}

// 16-bit A/B fragment load, matching CDNA5 WMMA VGPR layout:
// lane = 16*hf + row ; lane holds matrix row `row`, K-chunks [hf*8, hf*8+8)
// and [16+hf*8, 16+hf*8+8).  Works for A (row = M row) and for B when loading
// rows of B^T (row = N column).  Works on global or LDS pointers.
__device__ __forceinline__ v16bf load_frag(const __bf16* base, int stride,
                                           int row, int hf) {
    const __bf16* p = base + row * stride + hf * 8;
    v8bf a0 = *(const v8bf*)(p);
    v8bf a1 = *(const v8bf*)(p + 16);
    v16bf r;
#pragma unroll
    for (int i = 0; i < 8; ++i) { r[i] = a0[i]; r[i + 8] = a1[i]; }
    return r;
}

__device__ __forceinline__ v8f v8f_zero() {
    v8f z;
#pragma unroll
    for (int i = 0; i < 8; ++i) z[i] = 0.0f;
    return z;
}

// 16-byte global -> LDS copy: async (ASYNCcnt-tracked, no VGPR round trip)
// when the toolchain exposes the gfx1250 builtin, else a plain staged copy.
__device__ __forceinline__ void copy16_g2l(const __bf16* g, __bf16* l) {
#if HAVE_ASYNC_LDS
    __builtin_amdgcn_global_load_async_to_lds_b128(
        (AS1 v4i*)g, (AS3 v4i*)l, /*imm offset=*/0, /*cpol=*/0);
#else
    *(v8bf*)l = *(const v8bf*)g;
#endif
}

__device__ __forceinline__ void wait_async_all() {
#if HAVE_ASYNC_LDS
#if __has_builtin(__builtin_amdgcn_s_wait_asynccnt)
    __builtin_amdgcn_s_wait_asynccnt(0);
#else
    asm volatile("s_wait_asynccnt 0x0" ::: "memory");
#endif
#endif
}

// ---------------------------------------------------------------------------
// fp32 -> bf16 conversion (vectorized, n % 4 == 0)
// ---------------------------------------------------------------------------
__global__ void cvt_f32_bf16(const float* __restrict__ in,
                             __bf16* __restrict__ out, int n4) {
    int i = blockIdx.x * blockDim.x + threadIdx.x;
    if (i < n4) {
        float4 v = ((const float4*)in)[i];
        v4bf o;
        o[0] = (__bf16)v.x; o[1] = (__bf16)v.y;
        o[2] = (__bf16)v.z; o[3] = (__bf16)v.w;
        ((v4bf*)out)[i] = o;
    }
}

// ---------------------------------------------------------------------------
// GEMM:  C(MxN) = A(MxK) @ W^T,  W is (NxK) row-major (so W rows == B^T rows)
// Block tile 128x128, 8 waves (2x4), wave tile 64x32, K-step 32.
// A/W tiles are staged global->LDS with async copies, double buffered;
// all fragment traffic comes out of LDS (16B-padded rows, conflict-free).
// mode 0: store fp32 to out[m*N+n]                (final Wo projection)
// mode 1: store bf16 packed (B,H,Ntok,Dh)         (Q, K)
// mode 2: store bf16 packed (B,H,Dh,Ntok)         (V transposed)
// ---------------------------------------------------------------------------
#define KSTEP 32
#define LDSK  40   // 32 data + 8 pad elements per row (80B row stride)

__launch_bounds__(256)
__global__ void gemm_bf16(const __bf16* __restrict__ A,
                          const __bf16* __restrict__ W,
                          void* __restrict__ out,
                          int M, int N, int K, int mode) {
    __shared__ __bf16 sA[2][128 * LDSK];
    __shared__ __bf16 sB[2][128 * LDSK];

    const int tid  = threadIdx.x;
    const int lane = tid & 31;
    const int wv   = tid >> 5;
    const int hf   = lane >> 4;
    const int row  = lane & 15;
    const int wm   = wv >> 2;       // 0..1
    const int wn   = wv & 3;        // 0..3
    const int mb0  = blockIdx.y * 128;
    const int nb0  = blockIdx.x * 128;

    v8f acc[4][2];
#pragma unroll
    for (int i = 0; i < 4; ++i)
#pragma unroll
        for (int j = 0; j < 2; ++j) acc[i][j] = v8f_zero();

    // Stage one 128x32 tile of A and of W into LDS buffer `buf`.
    // 512 16B-chunks per matrix, 2 per thread per matrix.
    auto stage = [&](int buf, int kk) {
#pragma unroll
        for (int t = 0; t < 2; ++t) {
            int c   = tid + t * 256;
            int r   = c >> 2;
            int col = (c & 3) * 8;
            copy16_g2l(A + (size_t)(mb0 + r) * K + kk + col,
                       &sA[buf][r * LDSK + col]);
            copy16_g2l(W + (size_t)(nb0 + r) * K + kk + col,
                       &sB[buf][r * LDSK + col]);
        }
    };

    stage(0, 0);
    int ib = 0;
    for (int kk = 0; kk < K; kk += KSTEP, ib ^= 1) {
        wait_async_all();          // this wave's copies into buf `ib` done
        __syncthreads();           // everyone's copies visible, prev reads done

        if (kk + KSTEP < K) {
            __builtin_prefetch((const void*)(A + (size_t)mb0 * K + kk + 2 * KSTEP), 0, 3);
            __builtin_prefetch((const void*)(W + (size_t)nb0 * K + kk + 2 * KSTEP), 0, 3);
            stage(ib ^ 1, kk + KSTEP);
        }

        const __bf16* bufA = &sA[ib][0];
        const __bf16* bufB = &sB[ib][0];
        v16bf a[4], b[2];
#pragma unroll
        for (int mt = 0; mt < 4; ++mt)
            a[mt] = load_frag(bufA + (wm * 64 + mt * 16) * LDSK, LDSK, row, hf);
#pragma unroll
        for (int nt = 0; nt < 2; ++nt)
            b[nt] = load_frag(bufB + (wn * 32 + nt * 16) * LDSK, LDSK, row, hf);

#pragma unroll
        for (int mt = 0; mt < 4; ++mt)
#pragma unroll
            for (int nt = 0; nt < 2; ++nt)
                acc[mt][nt] = wmma_bf16(a[mt], b[nt], acc[mt][nt]);

        __syncthreads();           // reads of buf `ib` done before it is reused
    }

    // Epilogue.  C element (r, lane): m_local = r + 8*hf, n_local = row.
    const int mb = mb0 + wm * 64;
    const int nb = nb0 + wn * 32;
#pragma unroll
    for (int mt = 0; mt < 4; ++mt) {
#pragma unroll
        for (int nt = 0; nt < 2; ++nt) {
#pragma unroll
            for (int r = 0; r < 8; ++r) {
                int m = mb + mt * 16 + r + 8 * hf;
                int n = nb + nt * 16 + row;
                float v = acc[mt][nt][r];
                if (mode == 0) {
                    ((float*)out)[(size_t)m * N + n] = v;
                } else {
                    int bi = m >> 11, tn = m & 2047;   // batch, token (Ntok=2048)
                    int h  = n >> 6,  dh = n & 63;     // head, head-dim (Dh=64)
                    __bf16 bv = (__bf16)v;
                    if (mode == 1)
                        ((__bf16*)out)[(((size_t)(bi * 16 + h)) * 2048 + tn) * 64 + dh] = bv;
                    else
                        ((__bf16*)out)[(((size_t)(bi * 16 + h)) * 64 + dh) * 2048 + tn] = bv;
                }
            }
        }
    }
}

// ---------------------------------------------------------------------------
// RMSNorm over head_dim=64, in place on packed (B*H, N, 64) bf16.
// One wave per row; each lane owns 2 elements.
// ---------------------------------------------------------------------------
__global__ void rmsnorm_bf16(__bf16* __restrict__ X,
                             const float* __restrict__ w, int rows) {
    int gid  = blockIdx.x * blockDim.x + threadIdx.x;
    int wid  = gid >> 5;
    int lane = threadIdx.x & 31;
    if (wid >= rows) return;
    __bf16* p = X + (size_t)wid * 64;
    float x0 = (float)p[lane * 2];
    float x1 = (float)p[lane * 2 + 1];
    float ss = x0 * x0 + x1 * x1;
#pragma unroll
    for (int off = 16; off; off >>= 1) ss += __shfl_xor(ss, off, 32);
    float s = rsqrtf(ss * (1.0f / 64.0f) + 1.1920929e-07f);
    p[lane * 2]     = (__bf16)(x0 * s * w[lane * 2]);
    p[lane * 2 + 1] = (__bf16)(x1 * s * w[lane * 2 + 1]);
}

// ---------------------------------------------------------------------------
// Flash attention.  grid = (B*H, N/64), block = 128 (4 waves).
// Wave handles 16 queries, streams keys in blocks of 32, online softmax.
// Qp,Kp: (B,H,N,Dh) bf16 ; Vt: (B,H,Dh,N) bf16 ; AO: (B,N,D) bf16.
// Per-(b,h) K/V is 0.5 MB bf16, so cross-query-tile re-reads stay in L2.
// ---------------------------------------------------------------------------
__launch_bounds__(128)
__global__ void attn_flash(const __bf16* __restrict__ Qp,
                           const __bf16* __restrict__ Kp,
                           const __bf16* __restrict__ Vt,
                           __bf16* __restrict__ AO) {
    const int N = 2048, Dh = 64;
    const int bh   = blockIdx.x;            // b*16 + h
    const int q0   = blockIdx.y * 64;
    const int lane = threadIdx.x & 31;
    const int wv   = threadIdx.x >> 5;      // 0..3
    const int hf   = lane >> 4;
    const int row  = lane & 15;
    const int qw   = q0 + wv * 16;

    __shared__ __bf16 ldsP[4][16 * 32];     // per-wave P tile (16 q x 32 keys)

    const __bf16* Qb = Qp + (size_t)bh * N * Dh;
    const __bf16* Kb = Kp + (size_t)bh * N * Dh;
    const __bf16* Vb = Vt + (size_t)bh * Dh * N;

    v16bf aQ[2];
#pragma unroll
    for (int kc = 0; kc < 2; ++kc)
        aQ[kc] = load_frag(Qb + (size_t)qw * Dh + kc * 32, Dh, row, hf);

    v8f O[4];
#pragma unroll
    for (int d = 0; d < 4; ++d) O[d] = v8f_zero();
    float mrow[8], lrow[8];
#pragma unroll
    for (int r = 0; r < 8; ++r) { mrow[r] = -1e30f; lrow[r] = 0.0f; }

    for (int kb = 0; kb < N; kb += 32) {
        // S = Q @ K^T for 32 keys (two 16-key n-tiles, Dh split into 2 chunks)
        v16bf bK[2][2];
#pragma unroll
        for (int nt = 0; nt < 2; ++nt)
#pragma unroll
            for (int kc = 0; kc < 2; ++kc)
                bK[nt][kc] = load_frag(Kb + (size_t)(kb + nt * 16) * Dh + kc * 32,
                                       Dh, row, hf);
        v8f S0 = v8f_zero(), S1 = v8f_zero();
        S0 = wmma_bf16(aQ[0], bK[0][0], S0);
        S0 = wmma_bf16(aQ[1], bK[0][1], S0);
        S1 = wmma_bf16(aQ[0], bK[1][0], S1);
        S1 = wmma_bf16(aQ[1], bK[1][1], S1);

        // online softmax; row m = r + 8*hf lives in lanes {16*hf .. 16*hf+15}
#pragma unroll
        for (int r = 0; r < 8; ++r) {
            float s0 = S0[r] * 0.125f;       // scale = Dh^-0.5
            float s1 = S1[r] * 0.125f;
            float mx = fmaxf(s0, s1);
#pragma unroll
            for (int off = 8; off; off >>= 1)
                mx = fmaxf(mx, __shfl_xor(mx, off, 32));
            float mn = fmaxf(mrow[r], mx);
            float f  = __expf(mrow[r] - mn);
            float p0 = __expf(s0 - mn);
            float p1 = __expf(s1 - mn);
            float rs = p0 + p1;
#pragma unroll
            for (int off = 8; off; off >>= 1)
                rs += __shfl_xor(rs, off, 32);
            lrow[r] = lrow[r] * f + rs;
            mrow[r] = mn;
#pragma unroll
            for (int d = 0; d < 4; ++d) O[d][r] *= f;

            int m = r + 8 * hf;
            ldsP[wv][m * 32 + row]      = (__bf16)p0;
            ldsP[wv][m * 32 + 16 + row] = (__bf16)p1;
        }

        // per-wave LDS RAW: wait for ds stores before reloading as A-frag
        asm volatile("s_wait_dscnt 0x0" ::: "memory");
        v16bf aP = load_frag(&ldsP[wv][0], 32, row, hf);

        // O += P @ V  (B^T rows == Vt rows; one WMMA per 16-wide Dh tile)
#pragma unroll
        for (int d = 0; d < 4; ++d) {
            v16bf bV = load_frag(Vb + (size_t)(d * 16) * N + kb, N, row, hf);
            O[d] = wmma_bf16(aP, bV, O[d]);
        }
    }

    // epilogue: AO[(b*2048 + q) * 1024 + h*64 + d*16 + n]
    int bb = bh >> 4, h = bh & 15;
#pragma unroll
    for (int d = 0; d < 4; ++d) {
#pragma unroll
        for (int r = 0; r < 8; ++r) {
            int m = r + 8 * hf;
            int q = qw + m;
            float inv = 1.0f / lrow[r];
            AO[((size_t)(bb * 2048 + q)) * 1024 + h * 64 + d * 16 + row] =
                (__bf16)(O[d][r] * inv);
        }
    }
}

// ---------------------------------------------------------------------------
// Launch
// ---------------------------------------------------------------------------
extern "C" void kernel_launch(void* const* d_in, const int* in_sizes, int n_in,
                              void* d_out, int out_size, void* d_ws, size_t ws_size,
                              hipStream_t stream) {
    const float* x    = (const float*)d_in[0];
    const float* Wq   = (const float*)d_in[1];
    const float* Wk   = (const float*)d_in[2];
    const float* Wv   = (const float*)d_in[3];
    const float* Wo   = (const float*)d_in[4];
    const float* qn_w = (const float*)d_in[5];
    const float* kn_w = (const float*)d_in[6];
    float* out = (float*)d_out;

    const int B = 4, Ntok = 2048, D = 1024, H = 16, Dh = 64;
    const int M = B * Ntok;              // 8192 token rows
    const size_t xN = (size_t)M * D;     // 8388608
    const size_t wN = (size_t)D * D;     // 1048576
    const size_t pN = (size_t)B * H * Ntok * Dh;  // packed q/k/v elems

    char* ws = (char*)d_ws;
    __bf16* xb  = (__bf16*)ws;  ws += xN * 2;
    __bf16* Wqb = (__bf16*)ws;  ws += wN * 2;
    __bf16* Wkb = (__bf16*)ws;  ws += wN * 2;
    __bf16* Wvb = (__bf16*)ws;  ws += wN * 2;
    __bf16* Wob = (__bf16*)ws;  ws += wN * 2;
    __bf16* Qp  = (__bf16*)ws;  ws += pN * 2;
    __bf16* Kp  = (__bf16*)ws;  ws += pN * 2;
    __bf16* Vt  = (__bf16*)ws;  ws += pN * 2;
    __bf16* AOb = (__bf16*)ws;  ws += xN * 2;   // total ~88 MB

    // 1. fp32 -> bf16
    cvt_f32_bf16<<<(int)(xN / 4 / 256), 256, 0, stream>>>(x,  xb,  (int)(xN / 4));
    cvt_f32_bf16<<<(int)(wN / 4 / 256), 256, 0, stream>>>(Wq, Wqb, (int)(wN / 4));
    cvt_f32_bf16<<<(int)(wN / 4 / 256), 256, 0, stream>>>(Wk, Wkb, (int)(wN / 4));
    cvt_f32_bf16<<<(int)(wN / 4 / 256), 256, 0, stream>>>(Wv, Wvb, (int)(wN / 4));
    cvt_f32_bf16<<<(int)(wN / 4 / 256), 256, 0, stream>>>(Wo, Wob, (int)(wN / 4));

    // 2. QKV projections (packed epilogues)
    dim3 gg(D / 128, M / 128);           // (8, 64)
    gemm_bf16<<<gg, 256, 0, stream>>>(xb, Wqb, Qp, M, D, D, 1);
    gemm_bf16<<<gg, 256, 0, stream>>>(xb, Wkb, Kp, M, D, D, 1);
    gemm_bf16<<<gg, 256, 0, stream>>>(xb, Wvb, Vt, M, D, D, 2);

    // 3. RMSNorm on Q and K (in place; regenerated by GEMM each call)
    int rows = B * H * Ntok;             // 131072
    rmsnorm_bf16<<<rows * 32 / 256, 256, 0, stream>>>(Qp, qn_w, rows);
    rmsnorm_bf16<<<rows * 32 / 256, 256, 0, stream>>>(Kp, kn_w, rows);

    // 4. flash attention
    attn_flash<<<dim3(B * H, Ntok / 64), 128, 0, stream>>>(Qp, Kp, Vt, AOb);

    // 5. output projection (fp32 out)
    gemm_bf16<<<gg, 256, 0, stream>>>(AOb, Wob, out, M, D, D, 0);
}
